// MultiHeadAttention_40286793236532
// MI455X (gfx1250) — matrix operand order repaired and checked by reading, not compiled, not run on hardware
//
#include <hip/hip_runtime.h>
#include <hip/hip_bf16.h>

typedef __attribute__((ext_vector_type(16))) _Float16 v16h;
typedef __attribute__((ext_vector_type(8)))  float    v8f;
typedef __attribute__((ext_vector_type(2)))  float    v2f;
typedef __attribute__((ext_vector_type(4)))  unsigned int v4u;
typedef __attribute__((ext_vector_type(8)))  int      v8i;
typedef __attribute__((ext_vector_type(4)))  int      v4i;

#define D_MODEL 1024
#define N_SEQ   2048
#define N_BATCH 8
#define N_QKV   3072
#define M_TOT   (N_BATCH * N_SEQ)   // 16384

#ifndef __has_builtin
#define __has_builtin(x) 0
#endif
#if __has_builtin(__builtin_amdgcn_tensor_load_to_lds)
#define HAVE_TDM 1
#else
#define HAVE_TDM 0
#endif

// ---- gfx1250 async copy (ASYNCcnt) ----------------------------------------
__device__ __forceinline__ void async_copy_b128(unsigned ldsoff, const float* gp) {
    asm volatile("global_load_async_to_lds_b128 %0, %1, off"
                 :: "v"(ldsoff), "v"(gp) : "memory");
}
__device__ __forceinline__ void async_wait0() {
    asm volatile("s_wait_asynccnt 0" ::: "memory");
}
__device__ __forceinline__ void tensor_wait0() {
#if __has_builtin(__builtin_amdgcn_s_wait_tensorcnt)
    __builtin_amdgcn_s_wait_tensorcnt(0);
#else
    asm volatile("s_wait_tensorcnt 0" ::: "memory");
#endif
}
__device__ __forceinline__ unsigned lds_off(const void* p) {
    return (unsigned)(unsigned long long)p;   // LDS aperture: addr[31:0] = LDS byte address
}

#if HAVE_TDM
// TDM: load a 2D tile (tile_dim0=1024 f32 x tile_dim1=16 rows, row stride 3072
// elements) from global into contiguous LDS.  D# per CDNA5 ISA ch.8.
// clang-23 builtin arity: (v4u g0, v8i g1, v4i g2, v4i g3, v8i g4, i32 cpol).
__device__ __forceinline__ void tdm_load_16x1024(const float* gbase, unsigned ldsaddr) {
    const unsigned long long ga = (unsigned long long)gbase;
    v4u g0 = { 1u,                                    // count=1 (valid user D#)
               ldsaddr,                               // lds_addr (bytes)
               (unsigned)ga,                          // global_addr[31:0]
               ((unsigned)(ga >> 32) & 0x01FFFFFFu)   // global_addr[56:32]
                   | 0x80000000u };                   // type=2 ("image")
    v8i g1 = { (int)0x00020000,            // workgroup_mask=0 | data_size=2 (4B)
               (int)(1024u << 16),         // tensor_dim0[15:0] @ bit48
               (int)(16u << 16),           // tensor_dim0[31:16]=0 | tensor_dim1=16
               (int)(1024u << 16),         // tensor_dim1[31:16]=0 | tile_dim0=1024
               (int)16,                    // tile_dim1=16 | tile_dim2=0
               (int)3072,                  // tensor_dim0_stride[31:0]
               0, 0 };                     // stride hi | tensor_dim1_stride=0
    const v4i z4 = { 0, 0, 0, 0 };         // groups 2/3: dims 2..4 unused
    const v8i z8 = { 0, 0, 0, 0, 0, 0, 0, 0 };
    __builtin_amdgcn_tensor_load_to_lds(g0, g1, z4, z4, z8, 0);
}
#endif

// ---------------------------------------------------------------------------
// Kernel 1: mixed-precision QKV GEMM, 8-wave WG, 64x32 macro-tile.
//   K in [256,768):              v_wmma_f32_16x16x4_f32
//   K in [0,256) u [768,1024):   v_wmma_f32_16x16x32_f16 (cvt in-reg)
// A/B panels async-staged to LDS (rows padded to 36 floats = 144B).
// ---------------------------------------------------------------------------
#define APAD 36
__global__ __launch_bounds__(256)
void qkv_kernel(const float* __restrict__ x, const float* __restrict__ w,
                float* __restrict__ qkv) {
    const int row0 = blockIdx.x * 64;
    const int col0 = blockIdx.y * 32;
    const int t    = threadIdx.x;
    const int lane = t & 31, wave = t >> 5;
    const int hi = lane >> 4, lo = lane & 15;
    const int tr = wave & 3, tc = wave >> 2;

    __shared__ float A_lds[64 * APAD];
    __shared__ float B_lds[32 * APAD];

    v8f acc = {};

    for (int k0 = 0; k0 < D_MODEL; k0 += 32) {
#pragma unroll
        for (int i = 0; i < 2; ++i) {          // A: 64 rows x 32 f32
            const int c = t + i * 256;
            const int r = c >> 3, off = (c & 7) * 4;
            async_copy_b128(lds_off(&A_lds[r * APAD + off]),
                            x + (size_t)(row0 + r) * D_MODEL + k0 + off);
        }
        {                                      // B: 32 K-rows x 32 cols
            const int r = t >> 3, off = (t & 7) * 4;
            async_copy_b128(lds_off(&B_lds[r * APAD + off]),
                            w + (size_t)(k0 + r) * N_QKV + col0 + off);
        }
        async_wait0();
        __syncthreads();

        const float* Ar = &A_lds[(tr * 16 + lo) * APAD];
        const int co = tc * 16 + lo;

        if (k0 >= 256 && k0 < 768) {           // fp32 band (uniform branch)
#pragma unroll
            for (int kc = 0; kc < 8; ++kc) {
                const int kk = kc * 4 + hi * 2;
                v2f a;  a.x  = Ar[kk];                 a.y  = Ar[kk + 1];
                v2f bb; bb.x = B_lds[kk * APAD + co];  bb.y = B_lds[(kk + 1) * APAD + co];
                acc = __builtin_amdgcn_wmma_f32_16x16x4_f32(false, a, false, bb, (short)0, acc, false, false);
            }
        } else {                               // fp16 bands
            v16h a;
#pragma unroll
            for (int i = 0; i < 8; ++i) {
                a[i]     = (_Float16)Ar[hi * 8 + i];
                a[8 + i] = (_Float16)Ar[16 + hi * 8 + i];
            }
            v16h bh;
#pragma unroll
            for (int i = 0; i < 16; ++i)
                bh[i] = (_Float16)B_lds[((i < 8 ? 0 : 8) + hi * 8 + i) * APAD + co];
            acc = __builtin_amdgcn_wmma_f32_16x16x32_f16(false, a, false, bh, (short)0, acc, false, false);
        }
        __syncthreads();
    }

#pragma unroll
    for (int r = 0; r < 8; ++r)
        qkv[(size_t)(row0 + tr * 16 + r + 8 * hi) * N_QKV + col0 + tc * 16 + lo] = acc[r];
}

// ---------------------------------------------------------------------------
// Kernel 2: flash-attention, 8 waves/block; wave w owns d-slice w*128..+128.
// K tiles: double-buffered async-B128 into padded LDS (prefetch kt+1 overlaps
// compute of kt).  V tile: TDM tensor_load_to_lds (contiguous LDS; fragment
// reads go along rows -> conflict-free), waited with s_wait_tensorcnt.
// ---------------------------------------------------------------------------
#define KVP 1028                       // padded K row: 4112B, 16B-aligned
__global__ __launch_bounds__(256)
void attn_kernel(const float* __restrict__ qkv, float* __restrict__ y) {
    const int qt   = blockIdx.x;
    const int b    = blockIdx.y;
    const int lane = threadIdx.x & 31;
    const int wave = threadIdx.x >> 5;
    const int hi = lane >> 4, lo = lane & 15;
    const int dlo = wave * 128;

    __shared__ float Kb[2][16 * KVP];  // 2 x 65.8 KB
    __shared__ float Vt[16 * 1024];    // 64 KB, contiguous (TDM target)
    __shared__ float S_lds[16 * 16];
    __shared__ float m_lds[16], l_lds[16], a_lds[16];

    const long long qrow0 = (long long)(b * N_SEQ + qt * 16);

    auto issue_K = [&](int kt_, float* dst) {
#pragma unroll
        for (int i = 0; i < 16; ++i) {
            const int c = threadIdx.x + i * 256;
            const int r = c >> 8, off = (c & 255) * 4;
            async_copy_b128(lds_off(&dst[r * KVP + off]),
                            qkv + (size_t)(b * N_SEQ + kt_ * 16 + r) * N_QKV + D_MODEL + off);
        }
    };

    // Q A-fragments, pre-scaled by 1/sqrt(d) = 1/32
    v2f aQ[32];
#pragma unroll
    for (int kc = 0; kc < 32; ++kc) {
        const int kk = dlo + kc * 4 + hi * 2;
        const float* qp = qkv + (qrow0 + lo) * N_QKV + kk;
        aQ[kc].x = qp[0] * 0.03125f;
        aQ[kc].y = qp[1] * 0.03125f;
    }

    v8f acc[8];
#pragma unroll
    for (int nt = 0; nt < 8; ++nt) acc[nt] = (v8f){};

    issue_K(0, Kb[0]);                         // prologue prefetch
    if (threadIdx.x < 16) { m_lds[threadIdx.x] = -3.0e38f; l_lds[threadIdx.x] = 0.0f; }
    __syncthreads();

    for (int kt = 0; kt < 128; ++kt) {
        const int cur = kt & 1;
        const float* vsrc = qkv + (size_t)(b * N_SEQ + kt * 16) * N_QKV + 2 * D_MODEL;

        // V tile for this iteration (previous Vt readers retired at end barrier)
#if HAVE_TDM
        if (wave == 0) tdm_load_16x1024(vsrc, lds_off(Vt));
#else
#pragma unroll
        for (int i = 0; i < 16; ++i) {
            const int c = threadIdx.x + i * 256;
            const int r = c >> 8, off = (c & 255) * 4;
            async_copy_b128(lds_off(&Vt[r * 1024 + off]), vsrc + (size_t)r * N_QKV + off);
        }
#endif
        S_lds[threadIdx.x] = 0.0f;

        async_wait0();                         // K[cur] (and fallback V) arrived
        __syncthreads();

        if (kt + 1 < 128) issue_K(kt + 1, Kb[cur ^ 1]);   // overlap next K with compute

        // partial S over this wave's 128-wide slice (K fragments from LDS)
        v8f s = {};
        const float* kr = &Kb[cur][lo * KVP];
#pragma unroll
        for (int kc = 0; kc < 32; ++kc) {
            const int kk = dlo + kc * 4 + hi * 2;
            v2f bK; bK.x = kr[kk]; bK.y = kr[kk + 1];
            s = __builtin_amdgcn_wmma_f32_16x16x4_f32(false, aQ[kc], false, bK, (short)0, s, false, false);
        }
#pragma unroll
        for (int r = 0; r < 8; ++r)
            atomicAdd(&S_lds[(r + 8 * hi) * 16 + lo], s[r]);
        __syncthreads();

        // online softmax (lanes 0..15 of wave 0); wave 0 also drains the V TDM
        if (threadIdx.x < 16) {
            const int row = threadIdx.x;
            float mx = S_lds[row * 16];
#pragma unroll
            for (int j = 1; j < 16; ++j) mx = fmaxf(mx, S_lds[row * 16 + j]);
            const float m_old = m_lds[row];
            const float m_new = fmaxf(m_old, mx);
            const float alpha = __expf(m_old - m_new);
            float sum = 0.0f;
#pragma unroll
            for (int j = 0; j < 16; ++j) {
                const float p = __expf(S_lds[row * 16 + j] - m_new);
                S_lds[row * 16 + j] = p;
                sum += p;
            }
            l_lds[row] = l_lds[row] * alpha + sum;
            m_lds[row] = m_new;
            a_lds[row] = alpha;
        }
#if HAVE_TDM
        if (wave == 0) tensor_wait0();         // V in LDS before release barrier
#endif
        __syncthreads();

        // rescale accumulators by alpha(row)
        float alf[8];
#pragma unroll
        for (int r = 0; r < 8; ++r) alf[r] = a_lds[r + 8 * hi];
#pragma unroll
        for (int nt = 0; nt < 8; ++nt)
#pragma unroll
            for (int r = 0; r < 8; ++r) acc[nt][r] *= alf[r];

        // P as A-fragments
        v2f aP[4];
#pragma unroll
        for (int kc = 0; kc < 4; ++kc) {
            const int kk = kc * 4 + hi * 2;
            aP[kc].x = S_lds[lo * 16 + kk];
            aP[kc].y = S_lds[lo * 16 + kk + 1];
        }

        // acc += P * V (V fragments from contiguous LDS rows: conflict-free)
#pragma unroll
        for (int nt = 0; nt < 8; ++nt) {
            const int n = dlo + nt * 16 + lo;
#pragma unroll
            for (int kc = 0; kc < 4; ++kc) {
                const int kvr = kc * 4 + hi * 2;
                v2f bV; bV.x = Vt[kvr * 1024 + n]; bV.y = Vt[(kvr + 1) * 1024 + n];
                acc[nt] = __builtin_amdgcn_wmma_f32_16x16x4_f32(false, aP[kc], false, bV, (short)0, acc[nt], false, false);
            }
        }
        __syncthreads();                       // Vt/S_lds readers retired
    }

    float linv[8];
#pragma unroll
    for (int r = 0; r < 8; ++r) linv[r] = 1.0f / l_lds[r + 8 * hi];
#pragma unroll
    for (int nt = 0; nt < 8; ++nt) {
#pragma unroll
        for (int r = 0; r < 8; ++r) {
            const int row = r + 8 * hi;
            const int col = dlo + nt * 16 + lo;
            y[(qrow0 + row) * D_MODEL + col] = acc[nt][r] * linv[r];
        }
    }
}

// ---------------------------------------------------------------------------
// Kernel 3: out = y @ out_w.T + out_b.  Same 8-wave 64x32 macro-tile + async
// LDS staging as kernel 1 (fp32 only).  B panel rows are out_w rows
// (contiguous in k) -> fully coalesced staging.
// ---------------------------------------------------------------------------
__global__ __launch_bounds__(256)
void proj_kernel(const float* __restrict__ y, const float* __restrict__ ow,
                 const float* __restrict__ ob, float* __restrict__ out) {
    const int row0 = blockIdx.x * 64;
    const int col0 = blockIdx.y * 32;
    const int t    = threadIdx.x;
    const int lane = t & 31, wave = t >> 5;
    const int hi = lane >> 4, lo = lane & 15;
    const int tr = wave & 3, tc = wave >> 2;

    __shared__ float A_lds[64 * APAD];
    __shared__ float B_lds[32 * APAD];   // B_lds[n_local][k_local]

    v8f acc = {};

    for (int k0 = 0; k0 < D_MODEL; k0 += 32) {
#pragma unroll
        for (int i = 0; i < 2; ++i) {          // A: 64 rows of y
            const int c = t + i * 256;
            const int r = c >> 3, off = (c & 7) * 4;
            async_copy_b128(lds_off(&A_lds[r * APAD + off]),
                            y + (size_t)(row0 + r) * D_MODEL + k0 + off);
        }
        {                                      // B: 32 out_w rows (n), 32 k's each
            const int r = t >> 3, off = (t & 7) * 4;
            async_copy_b128(lds_off(&B_lds[r * APAD + off]),
                            ow + (size_t)(col0 + r) * D_MODEL + k0 + off);
        }
        async_wait0();
        __syncthreads();

        const float* Ar = &A_lds[(tr * 16 + lo) * APAD];
        const float* Br = &B_lds[(tc * 16 + lo) * APAD];
#pragma unroll
        for (int kc = 0; kc < 8; ++kc) {
            const int kk = kc * 4 + hi * 2;
            v2f a;  a.x  = Ar[kk]; a.y  = Ar[kk + 1];
            v2f bb; bb.x = Br[kk]; bb.y = Br[kk + 1];
            acc = __builtin_amdgcn_wmma_f32_16x16x4_f32(false, a, false, bb, (short)0, acc, false, false);
        }
        __syncthreads();
    }

    const float bias = ob[col0 + tc * 16 + lo];
#pragma unroll
    for (int r = 0; r < 8; ++r)
        out[(size_t)(row0 + tr * 16 + r + 8 * hi) * D_MODEL + col0 + tc * 16 + lo] = acc[r] + bias;
}

// ---------------------------------------------------------------------------
extern "C" void kernel_launch(void* const* d_in, const int* in_sizes, int n_in,
                              void* d_out, int out_size, void* d_ws, size_t ws_size,
                              hipStream_t stream) {
    const float* x    = (const float*)d_in[0];   // [8,2048,1024]
    const float* wqkv = (const float*)d_in[1];   // [1024,3072]
    const float* outw = (const float*)d_in[2];   // [1024,1024]
    const float* outb = (const float*)d_in[3];   // [1024]
    float* out = (float*)d_out;                  // [8,2048,1024]

    float* qkv = (float*)d_ws;                   // 192 MB (fits the 192 MB L2)
    float* yws = qkv + (size_t)M_TOT * N_QKV;    // 64 MB

    qkv_kernel<<<dim3(M_TOT / 64, N_QKV / 32), 256, 0, stream>>>(x, wqkv, qkv);
    attn_kernel<<<dim3(N_SEQ / 16, N_BATCH), 256, 0, stream>>>(qkv, yws);
    proj_kernel<<<dim3(M_TOT / 64, D_MODEL / 32), 256, 0, stream>>>(yws, outw, outb, out);
}
